// FDGRPretrainedModel_3942779978211
// MI455X (gfx1250) — matrix operand order repaired and checked by reading, not compiled, not run on hardware
//
#include <hip/hip_runtime.h>
#include <hip/hip_bf16.h>

typedef __attribute__((ext_vector_type(16))) _Float16 v16h;
typedef __attribute__((ext_vector_type(8)))  _Float16 v8h;
typedef __attribute__((ext_vector_type(8)))  float    v8f;

#define NTOT 1024   // 2*B*S rows
#define NHALF 512   // B*S
#define HDIM 768
#define HD   256

__device__ __forceinline__ float block_sum(float v, float* red) {
  int tid = threadIdx.x;
  red[tid] = v; __syncthreads();
  #pragma unroll
  for (int s = 128; s > 0; s >>= 1) {
    if (tid < s) red[tid] += red[tid + s];
    __syncthreads();
  }
  float r = red[0];
  __syncthreads();
  return r;
}

__device__ __forceinline__ float softplusf(float x) {
  return x > 20.f ? x : log1pf(__expf(x));
}

__device__ __forceinline__ v16h load_afrag(const _Float16* p) {
  // ISA 7.12.2 16-bit A 16x32 layout: two aligned 16B loads, no converts
  v8h lo = *reinterpret_cast<const v8h*>(p);
  v8h hi = *reinterpret_cast<const v8h*>(p + 16);
  return __builtin_shufflevector(lo, hi,
      0, 1, 2, 3, 4, 5, 6, 7, 8, 9, 10, 11, 12, 13, 14, 15);
}

__device__ __forceinline__ v8f wmma_f16(v16h a, v16h b, v8f c) {
  return __builtin_amdgcn_wmma_f32_16x16x32_f16(
      false, a, false, b, (short)0, c, false, false);
}

// ---- packing kernels (once per launch; amortized over all M-tiles) ----

__global__ __launch_bounds__(256)
void pack_f16(const float* __restrict__ X, _Float16* __restrict__ Y, int n) {
  int i = blockIdx.x * 256 + threadIdx.x;
  if (i < n) Y[i] = (_Float16)X[i];
}

// W (K x N fp32, row-major) -> Wp[n*K + k] f16 (transposed, K-contiguous):
// a B-fragment becomes two 16B loads per lane (k = 8h..8h+7, 16+8h..16+8h+7).
__global__ __launch_bounds__(256)
void pack_w(const float* __restrict__ W, _Float16* __restrict__ Wp, int K, int N) {
  int i = blockIdx.x * 256 + threadIdx.x;
  if (i < K * N) {
    int n = i / K, k = i - n * K;
    Wp[i] = (_Float16)W[(size_t)k * N + n];
  }
}

// ---- WMMA GEMM ----
// C(M x N) = act(A(M x K) @ W + bias).  A: f16 row-major (lda halves).
// Wp: f16 packed [n*K + k].  C: fp32 (optional), Ch: f16 row-major copy (optional).
// Tile: 32(M) x 64(N) per wave, software-pipelined over K.
// grid = (N/64, M/32), block = 32 (one wave, EXEC all-ones as WMMA requires).
__global__ __launch_bounds__(32)
void gemm_wmma(const _Float16* __restrict__ A, int lda,
               const _Float16* __restrict__ Wp, int K,
               const float* __restrict__ bias,
               float* __restrict__ C, _Float16* __restrict__ Ch,
               int ldc, int act) {
  const int lane = threadIdx.x;
  const int m0   = blockIdx.y * 32;
  const int n0   = blockIdx.x * 64;
  const int mloc = lane & 15;
  const int hgrp = lane >> 4;       // 0: lanes 0-15, 1: lanes 16-31

  v8f acc[2][4] = {};

  const _Float16* arow0 = A + (size_t)(m0 + mloc) * lda + hgrp * 8;
  const _Float16* arow1 = arow0 + (size_t)16 * lda;
  const _Float16* wcol  = Wp + (size_t)(n0 + mloc) * K + hgrp * 8;
  const size_t wstride = (size_t)16 * K;

  // prologue: K-step 0 fragments
  v16h a0 = load_afrag(arow0);
  v16h a1 = load_afrag(arow1);
  v16h b0 = load_afrag(wcol);
  v16h b1 = load_afrag(wcol + wstride);
  v16h b2 = load_afrag(wcol + 2 * wstride);
  v16h b3 = load_afrag(wcol + 3 * wstride);

  #pragma unroll 2
  for (int k0 = 32; k0 < K; k0 += 32) {
    // issue next-iteration loads BEFORE consuming current fragments:
    // the matrix pipe executes the 8 WMMAs below while these are in flight.
    v16h na0 = load_afrag(arow0 + k0);
    v16h na1 = load_afrag(arow1 + k0);
    v16h nb0 = load_afrag(wcol + k0);
    v16h nb1 = load_afrag(wcol + wstride + k0);
    v16h nb2 = load_afrag(wcol + 2 * wstride + k0);
    v16h nb3 = load_afrag(wcol + 3 * wstride + k0);

    acc[0][0] = wmma_f16(a0, b0, acc[0][0]);
    acc[0][1] = wmma_f16(a0, b1, acc[0][1]);
    acc[0][2] = wmma_f16(a0, b2, acc[0][2]);
    acc[0][3] = wmma_f16(a0, b3, acc[0][3]);
    acc[1][0] = wmma_f16(a1, b0, acc[1][0]);
    acc[1][1] = wmma_f16(a1, b1, acc[1][1]);
    acc[1][2] = wmma_f16(a1, b2, acc[1][2]);
    acc[1][3] = wmma_f16(a1, b3, acc[1][3]);

    a0 = na0; a1 = na1; b0 = nb0; b1 = nb1; b2 = nb2; b3 = nb3;
  }
  // epilogue K-step
  acc[0][0] = wmma_f16(a0, b0, acc[0][0]);
  acc[0][1] = wmma_f16(a0, b1, acc[0][1]);
  acc[0][2] = wmma_f16(a0, b2, acc[0][2]);
  acc[0][3] = wmma_f16(a0, b3, acc[0][3]);
  acc[1][0] = wmma_f16(a1, b0, acc[1][0]);
  acc[1][1] = wmma_f16(a1, b1, acc[1][1]);
  acc[1][2] = wmma_f16(a1, b2, acc[1][2]);
  acc[1][3] = wmma_f16(a1, b3, acc[1][3]);

  // epilogue: C/D layout: VGPR i -> M = hgrp*8 + i, N = lane&15
  #pragma unroll
  for (int mt = 0; mt < 2; ++mt) {
    #pragma unroll
    for (int t = 0; t < 4; ++t) {
      const int col = n0 + t * 16 + mloc;
      const float bv = bias ? bias[col] : 0.f;
      #pragma unroll
      for (int i = 0; i < 8; ++i) {
        float v = acc[mt][t][i] + bv;
        if (act) v = fmaxf(v, 0.f);
        const size_t off = (size_t)(m0 + mt * 16 + hgrp * 8 + i) * ldc + col;
        if (C)  C[off]  = v;
        if (Ch) Ch[off] = (_Float16)v;
      }
    }
  }
}

__global__ __launch_bounds__(256)
void ln_kernel(const float* __restrict__ X, const float* __restrict__ g,
               const float* __restrict__ bt, float* __restrict__ Y,
               float* __restrict__ Y2, _Float16* __restrict__ Yh, int D) {
  __shared__ float red[256];
  const int row = blockIdx.x, tid = threadIdx.x;
  const float* x = X + (size_t)row * D;
  float s = 0.f;
  for (int i = tid; i < D; i += 256) s += x[i];
  const float mean = block_sum(s, red) / (float)D;
  float v = 0.f;
  for (int i = tid; i < D; i += 256) { float d = x[i] - mean; v += d * d; }
  const float var = block_sum(v, red) / (float)D;
  const float rstd = rsqrtf(var + 1e-12f);
  float*     y  = Y + (size_t)row * D;
  float*     y2 = Y2 ? Y2 + (size_t)row * D : nullptr;
  _Float16*  yh = Yh ? Yh + (size_t)row * D : nullptr;
  for (int i = tid; i < D; i += 256) {
    float val = (x[i] - mean) * rstd * g[i] + bt[i];
    y[i] = val;
    if (y2) y2[i] = val;
    if (yh) yh[i] = (_Float16)val;
  }
}

__global__ __launch_bounds__(256)
void concat_h_kernel(const _Float16* __restrict__ a, const _Float16* __restrict__ b,
                     _Float16* __restrict__ c) {
  int idx = blockIdx.x * 256 + threadIdx.x;     // < 1024*512
  int row = idx >> 9, col = idx & 511;
  c[idx] = (col < HD) ? a[(size_t)row * HD + col] : b[(size_t)row * HD + col - HD];
}

__global__ __launch_bounds__(256)
void sqdiff_partial(const float* __restrict__ a, const float* __restrict__ b,
                    int n, float* __restrict__ part) {
  __shared__ float red[256];
  float s = 0.f;
  for (int i = blockIdx.x * 256 + threadIdx.x; i < n; i += gridDim.x * 256) {
    float d = a[i] - b[i]; s += d * d;
  }
  float t = block_sum(s, red);
  if (threadIdx.x == 0) part[blockIdx.x] = t;
}

__global__ __launch_bounds__(256)
void t0_kernel(const float* __restrict__ Ph, const float* __restrict__ Qh,
               const float* __restrict__ b1, const float* __restrict__ w2,
               const float* __restrict__ b2, float* __restrict__ t0) {
  __shared__ float red[256];
  const int r = blockIdx.x, k = threadIdx.x;
  float h = fmaxf(Ph[(size_t)r * HD + k] + Qh[(size_t)r * HD + k] + b1[k], 0.f) * w2[k];
  float s = block_sum(h, red);
  if (k == 0) t0[r] = softplusf(s + b2[0]);
}

// One block per i (row of T1). T1[i,j] = softplus(sum_k relu(Ph[j,k]+Qh[i,k]+b1[k])*w2[k]+b2)
// masked to -1e9 where ids_c[i]==ids_o[j] && i!=j. Produces lse[i] via online logsumexp.
__global__ __launch_bounds__(256)
void lse_kernel(const float* __restrict__ Ph, const float* __restrict__ Qh,
                const float* __restrict__ b1, const float* __restrict__ w2,
                const float* __restrict__ b2,
                const int* __restrict__ ids_o, const int* __restrict__ ids_c,
                float* __restrict__ lse) {
  __shared__ float qb[256];
  __shared__ float w2s[256];
  __shared__ float sm[256];
  __shared__ float ss[256];
  const int i = blockIdx.x, tid = threadIdx.x;
  qb[tid]  = Qh[(size_t)i * HD + tid] + b1[tid];
  w2s[tid] = w2[tid];
  __syncthreads();
  const int wi = ids_c[i];
  const float b2v = b2[0];
  __builtin_prefetch(Ph + (size_t)(tid + 256) * HD, 0, 1);
  float m = -3.4e38f, s = 0.f;
  #pragma unroll
  for (int jj = 0; jj < 2; ++jj) {
    const int j = tid + jj * 256;
    float t;
    if (wi == ids_o[j] && j != i) {
      t = -1e9f;
    } else {
      const float* ph = Ph + (size_t)j * HD;
      float acc = 0.f;
      #pragma unroll 8
      for (int k = 0; k < HD; ++k) acc = fmaf(fmaxf(ph[k] + qb[k], 0.f), w2s[k], acc);
      t = softplusf(acc + b2v);
    }
    if (t > m) { s = s * __expf(m - t) + 1.f; m = t; }
    else       { s += __expf(t - m); }
  }
  sm[tid] = m; ss[tid] = s; __syncthreads();
  #pragma unroll
  for (int st = 128; st > 0; st >>= 1) {
    if (tid < st) {
      float m1 = sm[tid], s1 = ss[tid], m2 = sm[tid + st], s2 = ss[tid + st];
      float mm = fmaxf(m1, m2);
      ss[tid] = s1 * __expf(m1 - mm) + s2 * __expf(m2 - mm);
      sm[tid] = mm;
    }
    __syncthreads();
  }
  if (tid == 0) lse[i] = sm[0] + logf(ss[0]);
}

__global__ __launch_bounds__(256)
void vad_orth_kernel(const float* __restrict__ vbuf, const float* __restrict__ abuf,
                     const float* __restrict__ dbuf,
                     const float* __restrict__ pvw, const float* __restrict__ paw,
                     const float* __restrict__ pdw,
                     const float* __restrict__ pvb, const float* __restrict__ pab,
                     const float* __restrict__ pdb,
                     const float* __restrict__ vad_o, const float* __restrict__ vad_c,
                     float* __restrict__ vad_part, float* __restrict__ orth_part) {
  __shared__ float red[256];
  const int row = blockIdx.x, k = threadIdx.x;
  const size_t idx = (size_t)row * HD + k;
  const float v = vbuf[idx], a = abuf[idx], d = dbuf[idx];
  float pv  = block_sum(v * pvw[k], red);
  float pa  = block_sum(a * paw[k], red);
  float pd  = block_sum(d * pdw[k], red);
  float gvv = block_sum(v * v, red);
  float gva = block_sum(v * a, red);
  float gvd = block_sum(v * d, red);
  float gaa = block_sum(a * a, red);
  float gad = block_sum(a * d, red);
  float gdd = block_sum(d * d, red);
  if (k == 0) {
    pv += pvb[0]; pa += pab[0]; pd += pdb[0];
    const float* tv = (row < NHALF) ? vad_o + (size_t)row * 3
                                    : vad_c + (size_t)(row - NHALF) * 3;
    float e0 = pv - tv[0], e1 = pa - tv[1], e2 = pd - tv[2];
    vad_part[row] = e0 * e0 + e1 * e1 + e2 * e2;
    float o0 = gvv - 1.f, o1 = gaa - 1.f, o2 = gdd - 1.f;
    orth_part[row] = o0 * o0 + o1 * o1 + o2 * o2
                   + 2.f * (gva * gva + gvd * gvd + gad * gad);
  }
}

__global__ __launch_bounds__(256)
void finalize_kernel(const float* __restrict__ club_part, const float* __restrict__ rec_part,
                     const float* __restrict__ t0, const float* __restrict__ lse,
                     const float* __restrict__ vad_part, const float* __restrict__ orth_part,
                     float* __restrict__ out) {
  __shared__ float red[256];
  const int tid = threadIdx.x;
  float c = club_part[tid];
  float r = rec_part[tid];
  float t = 0.f, l = 0.f;
  for (int i = tid; i < NHALF; i += 256) { t += t0[i]; l += lse[i]; }
  float vd = 0.f, og = 0.f;
  for (int i = tid; i < NTOT; i += 256) { vd += vad_part[i]; og += orth_part[i]; }
  float cs = block_sum(c, red);
  float rs = block_sum(r, red);
  float ts = block_sum(t, red);
  float ls = block_sum(l, red);
  float vs = block_sum(vd, red);
  float gs = block_sum(og, red);
  if (tid == 0) {
    out[0] = 0.5f * cs / (float)NTOT;                              // club_loss
    out[1] = rs / (float)(NTOT * HDIM);                            // reconstruct_loss
    out[2] = gs;                                                   // orthogonal_loss
    out[3] = -(ts / (float)NHALF - (ls / (float)NHALF - logf((float)NHALF))); // ha_loss
    out[4] = vs / (float)(NTOT * 3);                               // vad_loss
  }
}

extern "C" void kernel_launch(void* const* d_in, const int* in_sizes, int n_in,
                              void* d_out, int out_size, void* d_ws, size_t ws_size,
                              hipStream_t stream) {
  (void)in_sizes; (void)n_in; (void)out_size; (void)ws_size;
  const float* seq   = (const float*)d_in[0];
  const float* vad_o = (const float*)d_in[1];
  const float* vad_c = (const float*)d_in[2];
  const int*   ids_o = (const int*)d_in[3];
  const int*   ids_c = (const int*)d_in[4];
  // d_in[5] = attention_mask (unused by the reference math)
  const float* ha_w1 = (const float*)d_in[6];
  const float* ha_w2 = (const float*)d_in[7];
  const float* hc_w1 = (const float*)d_in[8];
  const float* hc_w2 = (const float*)d_in[9];
  const float* de_w1 = (const float*)d_in[10];
  const float* de_w2 = (const float*)d_in[11];
  const float* v_w   = (const float*)d_in[12];
  const float* a_w   = (const float*)d_in[13];
  const float* d_w   = (const float*)d_in[14];
  const float* pv_w  = (const float*)d_in[15];
  const float* pa_w  = (const float*)d_in[16];
  const float* pd_w  = (const float*)d_in[17];
  const float* f_w1  = (const float*)d_in[18];
  const float* f_w2  = (const float*)d_in[19];
  const float* ha_b1 = (const float*)d_in[20];
  const float* ha_b2 = (const float*)d_in[21];
  const float* ha_g  = (const float*)d_in[22];
  const float* ha_bt = (const float*)d_in[23];
  const float* hc_b1 = (const float*)d_in[24];
  const float* hc_b2 = (const float*)d_in[25];
  const float* hc_g  = (const float*)d_in[26];
  const float* hc_bt = (const float*)d_in[27];
  const float* de_b1 = (const float*)d_in[28];
  const float* de_b2 = (const float*)d_in[29];
  const float* de_g  = (const float*)d_in[30];
  const float* de_bt = (const float*)d_in[31];
  const float* v_b   = (const float*)d_in[32];
  const float* a_b   = (const float*)d_in[33];
  const float* d_b   = (const float*)d_in[34];
  const float* pv_b  = (const float*)d_in[35];
  const float* pa_b  = (const float*)d_in[36];
  const float* pd_b  = (const float*)d_in[37];
  const float* f_b1  = (const float*)d_in[38];
  const float* f_b2  = (const float*)d_in[39];

  float* out = (float*)d_out;
  float* ha_out = out + 5;                 // (1024,256) flat
  float* hc_out = out + 5 + NTOT * HD;

  // ---- fp32 workspace ----
  float* ws = (float*)d_ws;
  float* h2        = ws;                    // 1024*768 pre-LN scratch (reused)
  float* deco      = ws + 786432;           // 1024*768
  float* Ph        = ws + 1572864;          // 512*256
  float* Qh        = ws + 1703936;          // 512*256
  float* vbuf      = ws + 1835008;          // 1024*256
  float* abuf      = ws + 2097152;
  float* dbuf      = ws + 2359296;
  float* ha_ws     = ws + 2621440;          // fp32 ha (club loss)
  float* hc_ws     = ws + 2883584;          // fp32 hc
  float* t0b       = ws + 3145728;          // 512
  float* lseb      = ws + 3146240;          // 512
  float* club_part = ws + 3146752;          // 256
  float* rec_part  = ws + 3147008;          // 256
  float* vad_part  = ws + 3147264;          // 1024
  float* orth_part = ws + 3148288;          // 1024

  // ---- f16 workspace (32B-aligned region) ----
  _Float16* hb = (_Float16*)(ws + 3149312);
  _Float16* seq_h  = hb;                    // 1024*768
  _Float16* h1h    = hb + 786432;           // 1024*768
  _Float16* cat_h  = hb + 1572864;          // 1024*512
  _Float16* dec1_h = hb + 2097152;          // 1024*768
  _Float16* ha_h   = hb + 2883584;          // 1024*256
  _Float16* hc_h   = hb + 3145728;          // 1024*256
  _Float16* ha_w1p = hb + 3407872;          // 768*768
  _Float16* ha_w2p = hb + 3997696;          // 768*256
  _Float16* hc_w1p = hb + 4194304;          // 768*768
  _Float16* hc_w2p = hb + 4784128;          // 768*256
  _Float16* de_w1p = hb + 4980736;          // 512*768
  _Float16* de_w2p = hb + 5373952;          // 768*768
  _Float16* fw1t_p = hb + 5963776;          // 256*256
  _Float16* fw1b_p = hb + 6029312;          // 256*256
  _Float16* v_wp   = hb + 6094848;          // 256*256
  _Float16* a_wp   = hb + 6160384;
  _Float16* d_wp   = hb + 6225920;

  // ---- pack inputs & weights into WMMA-fragment-friendly f16 ----
  pack_f16<<<(NTOT*HDIM)/256, 256, 0, stream>>>(seq, seq_h, NTOT*HDIM);
  pack_w<<<(HDIM*HDIM)/256, 256, 0, stream>>>(ha_w1, ha_w1p, HDIM, HDIM);
  pack_w<<<(HDIM*HD)/256,   256, 0, stream>>>(ha_w2, ha_w2p, HDIM, HD);
  pack_w<<<(HDIM*HDIM)/256, 256, 0, stream>>>(hc_w1, hc_w1p, HDIM, HDIM);
  pack_w<<<(HDIM*HD)/256,   256, 0, stream>>>(hc_w2, hc_w2p, HDIM, HD);
  pack_w<<<(2*HD*HDIM)/256, 256, 0, stream>>>(de_w1, de_w1p, 2*HD, HDIM);
  pack_w<<<(HDIM*HDIM)/256, 256, 0, stream>>>(de_w2, de_w2p, HDIM, HDIM);
  pack_w<<<(HD*HD)/256, 256, 0, stream>>>(f_w1,                   fw1t_p, HD, HD);
  pack_w<<<(HD*HD)/256, 256, 0, stream>>>(f_w1 + (size_t)HD*HD,   fw1b_p, HD, HD);
  pack_w<<<(HD*HD)/256, 256, 0, stream>>>(v_w, v_wp, HD, HD);
  pack_w<<<(HD*HD)/256, 256, 0, stream>>>(a_w, a_wp, HD, HD);
  pack_w<<<(HD*HD)/256, 256, 0, stream>>>(d_w, d_wp, HD, HD);

  // ---- encoder ha ----
  gemm_wmma<<<dim3(HDIM/64, NTOT/32), 32, 0, stream>>>(seq_h, HDIM, ha_w1p, HDIM, ha_b1, nullptr, h1h, HDIM, 1);
  gemm_wmma<<<dim3(HD/64,   NTOT/32), 32, 0, stream>>>(h1h, HDIM, ha_w2p, HDIM, ha_b2, h2, nullptr, HD, 1);
  ln_kernel<<<NTOT, 256, 0, stream>>>(h2, ha_g, ha_bt, ha_ws, ha_out, ha_h, HD);
  // ---- encoder hc ----
  gemm_wmma<<<dim3(HDIM/64, NTOT/32), 32, 0, stream>>>(seq_h, HDIM, hc_w1p, HDIM, hc_b1, nullptr, h1h, HDIM, 1);
  gemm_wmma<<<dim3(HD/64,   NTOT/32), 32, 0, stream>>>(h1h, HDIM, hc_w2p, HDIM, hc_b2, h2, nullptr, HD, 1);
  ln_kernel<<<NTOT, 256, 0, stream>>>(h2, hc_g, hc_bt, hc_ws, hc_out, hc_h, HD);
  // ---- decoder ----
  concat_h_kernel<<<(NTOT*2*HD)/256, 256, 0, stream>>>(ha_h, hc_h, cat_h);
  gemm_wmma<<<dim3(HDIM/64, NTOT/32), 32, 0, stream>>>(cat_h, 2*HD, de_w1p, 2*HD, de_b1, nullptr, dec1_h, HDIM, 1);
  gemm_wmma<<<dim3(HDIM/64, NTOT/32), 32, 0, stream>>>(dec1_h, HDIM, de_w2p, HDIM, de_b2, h2, nullptr, HDIM, 1);
  ln_kernel<<<NTOT, 256, 0, stream>>>(h2, de_g, de_bt, deco, nullptr, nullptr, HDIM);
  // ---- scalar-loss partials ----
  sqdiff_partial<<<256, 256, 0, stream>>>(ha_ws, hc_ws, NTOT * HD, club_part);
  sqdiff_partial<<<256, 256, 0, stream>>>(seq, deco, NTOT * HDIM, rec_part);
  // ---- CLUB critic: F's first layer factors -> Ph, Qh ----
  gemm_wmma<<<dim3(HD/64, NHALF/32), 32, 0, stream>>>(ha_h, HD, fw1t_p, HD, nullptr, Ph, nullptr, HD, 0);
  gemm_wmma<<<dim3(HD/64, NHALF/32), 32, 0, stream>>>(ha_h + (size_t)NHALF*HD, HD, fw1b_p, HD, nullptr, Qh, nullptr, HD, 0);
  t0_kernel<<<NHALF, 256, 0, stream>>>(Ph, Qh, f_b1, f_w2, f_b2, t0b);
  lse_kernel<<<NHALF, 256, 0, stream>>>(Ph, Qh, f_b1, f_w2, f_b2, ids_o, ids_c, lseb);
  // ---- VAD heads ----
  gemm_wmma<<<dim3(HD/64, NTOT/32), 32, 0, stream>>>(hc_h, HD, v_wp, HD, v_b, vbuf, nullptr, HD, 0);
  gemm_wmma<<<dim3(HD/64, NTOT/32), 32, 0, stream>>>(hc_h, HD, a_wp, HD, a_b, abuf, nullptr, HD, 0);
  gemm_wmma<<<dim3(HD/64, NTOT/32), 32, 0, stream>>>(hc_h, HD, d_wp, HD, d_b, dbuf, nullptr, HD, 0);
  vad_orth_kernel<<<NTOT, 256, 0, stream>>>(vbuf, abuf, dbuf, pv_w, pa_w, pd_w,
                                            pv_b, pa_b, pd_b, vad_o, vad_c,
                                            vad_part, orth_part);
  // ---- final losses ----
  finalize_kernel<<<1, 256, 0, stream>>>(club_part, rec_part, t0b, lseb,
                                         vad_part, orth_part, out);
}
